// MultiHeadAttention2D_42082089566458
// MI455X (gfx1250) — matrix operand order repaired and checked by reading, not compiled, not run on hardware
//
#include <hip/hip_runtime.h>

#define DD   1024
#define BB   512
#define SS   78
#define HH   16
#define DKK  64
#define SP   80          // padded S for Q/K tiles
#define SPV  96          // padded S for V^T (3 K-chunks of 32)
#define NTOK (BB * SS)   // 39936
#define BHN  (BB * HH)   // 8192

typedef __attribute__((ext_vector_type(16))) __bf16 bf16x16;
typedef __attribute__((ext_vector_type(8)))  __bf16 bf16x8;
typedef __attribute__((ext_vector_type(8)))  float  f32x8;
typedef __attribute__((ext_vector_type(4)))  float  f32x4;

#if defined(__gfx1250__) && defined(__has_builtin)
#if __has_builtin(__builtin_amdgcn_tensor_load_to_lds) && \
    __has_builtin(__builtin_amdgcn_s_wait_tensorcnt)
#define HAVE_TDM 1
#endif
#endif
#ifndef HAVE_TDM
#define HAVE_TDM 0
#endif

__device__ __forceinline__ f32x8 zero8() {
  f32x8 z;
#pragma unroll
  for (int i = 0; i < 8; ++i) z[i] = 0.0f;
  return z;
}

__device__ __forceinline__ f32x8 wmma_bf16(bf16x16 a, bf16x16 b, f32x8 c) {
  return __builtin_amdgcn_wmma_f32_16x16x32_bf16(false, a, false, b, (short)0, c,
                                                 false, false);
}

// A fragment (16x32 bf16), lane L: row L%16, K = {kb..kb+15}, kb = 16*(L/16)
__device__ __forceinline__ bf16x16 load_a_f32(const float* rowp, int kk, int lane) {
  int kb = kk + ((lane >> 4) << 4);
  f32x4 a0 = *(const f32x4*)(rowp + kb);
  f32x4 a1 = *(const f32x4*)(rowp + kb + 4);
  f32x4 a2 = *(const f32x4*)(rowp + kb + 8);
  f32x4 a3 = *(const f32x4*)(rowp + kb + 12);
  bf16x16 r;
#pragma unroll
  for (int i = 0; i < 4; ++i) {
    r[i]      = (__bf16)a0[i];
    r[i + 4]  = (__bf16)a1[i];
    r[i + 8]  = (__bf16)a2[i];
    r[i + 12] = (__bf16)a3[i];
  }
  return r;
}

__device__ __forceinline__ bf16x16 load_a_bf16(const __bf16* rowp, int kk, int lane) {
  int kb = kk + ((lane >> 4) << 4);
  bf16x8 lo = *(const bf16x8*)(rowp + kb);
  bf16x8 hi = *(const bf16x8*)(rowp + kb + 8);
  return __builtin_shufflevector(lo, hi, 0, 1, 2, 3, 4, 5, 6, 7,
                                 8, 9, 10, 11, 12, 13, 14, 15);
}

#if HAVE_TDM
typedef __attribute__((ext_vector_type(4))) unsigned int u32x4;
typedef __attribute__((ext_vector_type(8))) int          i32x8;
typedef __attribute__((ext_vector_type(4))) int          i32x4;

// 2D tile DMA: global (row-major, stride0 elems/row, bf16) -> contiguous LDS tile.
// Rows/cols beyond tensor_dim are zero-filled by the TDM (ISA 8.2).
__device__ __forceinline__ void tdm_load_2d(const __bf16* gsrc, unsigned lds_off,
                                            unsigned dim0, unsigned dim1,
                                            unsigned til0, unsigned til1,
                                            unsigned stride0) {
  unsigned long long ga = (unsigned long long)(uintptr_t)gsrc;
  u32x4 g0 = { 1u,                                   // count=1, user mode
               lds_off,                              // LDS byte address
               (unsigned)ga,                         // global_addr[31:0]
               (unsigned)((ga >> 32) & 0x01FFFFFFu) | (2u << 30) };  // [56:32] | type=2
  i32x8 g1 = { (int)(1u << 16),                      // data_size = 2 bytes
               (int)((dim0 & 0xFFFFu) << 16),        // tensor_dim0[15:0] @ bits 63:48
               (int)((dim0 >> 16) | ((dim1 & 0xFFFFu) << 16)),
               (int)((dim1 >> 16) | (til0 << 16)),   // tile_dim0 @ bits 127:112
               (int)(til1 & 0xFFFFu),                // tile_dim1 (tile_dim2=0)
               (int)stride0,                         // tensor_dim0_stride lo32
               0, 0 };
  i32x4 z4 = { 0, 0, 0, 0 };
  i32x8 z8 = { 0, 0, 0, 0, 0, 0, 0, 0 };
  __builtin_amdgcn_tensor_load_to_lds(g0, g1, z4, z4, z8, 0);
}
#endif

// ---------------------------------------------------------------- weights fp32->bf16
__global__ __launch_bounds__(256) void convert_weights_kernel(
    const float* __restrict__ Wq, const float* __restrict__ Wk,
    const float* __restrict__ Wv, const float* __restrict__ Wo,
    __bf16* __restrict__ dst) {
  int i = blockIdx.x * blockDim.x + threadIdx.x;   // 0 .. 4*2^20-1
  int which = i >> 20;
  int off   = i & ((1 << 20) - 1);
  const float* src = (which == 0) ? Wq : (which == 1) ? Wk : (which == 2) ? Wv : Wo;
  dst[i] = (__bf16)src[off];
}

// ---------------------------------------------------------------- bias -> padded [16][80][80]
__global__ __launch_bounds__(256) void pad_bias_kernel(const float* __restrict__ bias,
                                                       float* __restrict__ bias_p) {
  int i = blockIdx.x * blockDim.x + threadIdx.x;   // 16*80*80 = 102400
  if (i >= HH * SP * SP) return;
  int k = i % SP;
  int t = i / SP;
  int q = t % SP;
  int h = t / SP;
  bias_p[i] = (q < SS && k < SS) ? bias[((size_t)h * SS + q) * SS + k] : 0.0f;
}

// ---------------------------------------------------------------- GEMM: out = X @ W^T
// MODE 0: X fp32, store bf16 head-major [B,H,80,64]    (Q and K)
// MODE 1: X fp32, store bf16 transposed [B,H,64,96]    (V)
// MODE 2: X bf16 (ctx), store fp32 [N,1024]            (output projection)
template <int MODE>
__global__ __launch_bounds__(256) void proj_gemm(const float* __restrict__ Xf,
                                                 const __bf16* __restrict__ Xbf,
                                                 const __bf16* __restrict__ W,
                                                 __bf16* __restrict__ dst_bf,
                                                 float* __restrict__ dst_f) {
  const int lane = threadIdx.x & 31;
  const int wave = threadIdx.x >> 5;
  const int m0   = (blockIdx.y * 8 + wave) * 16;   // 16-row strip per wave
  const int n0   = blockIdx.x * 128;               // 128 output columns per block
  const int half = lane >> 4;
  const int nc   = lane & 15;

  f32x8 acc[8];
#pragma unroll
  for (int t = 0; t < 8; ++t) acc[t] = zero8();

  const float*  arpf = (MODE != 2) ? Xf  + (size_t)(m0 + nc) * DD : nullptr;
  const __bf16* arpb = (MODE == 2) ? Xbf + (size_t)(m0 + nc) * DD : nullptr;

  for (int kk = 0; kk < DD; kk += 32) {
    bf16x16 a;
    if constexpr (MODE == 2) {
      a = load_a_bf16(arpb, kk, lane);
    } else {
      a = load_a_f32(arpf, kk, lane);
      __builtin_prefetch(arpf + kk + 256, 0, 0);   // speculative, OOB-safe
    }
#pragma unroll
    for (int t = 0; t < 8; ++t) {
      // B fragment: lane holds W row (n0+16t+nc), contiguous K chunk of 16
      const __bf16* brp = W + (size_t)(n0 + t * 16 + nc) * DD + kk + (half << 4);
      bf16x16 b = *(const bf16x16*)brp;
      acc[t] = wmma_bf16(a, b, acc[t]);
    }
  }

#pragma unroll
  for (int t = 0; t < 8; ++t) {
#pragma unroll
    for (int r = 0; r < 8; ++r) {
      int m = m0 + half * 8 + r;     // token index
      int n = n0 + t * 16 + nc;      // output channel
      float v = acc[t][r];
      if constexpr (MODE == 2) {
        dst_f[(size_t)m * DD + n] = v;
      } else {
        int b  = m / SS, s = m % SS;
        int h  = n >> 6, dk = n & 63;
        if constexpr (MODE == 0)
          dst_bf[(((size_t)(b * HH + h)) * SP + s) * DKK + dk] = (__bf16)v;
        else
          dst_bf[(((size_t)(b * HH + h)) * DKK + dk) * SPV + s] = (__bf16)v;
      }
    }
  }
}

// ---------------------------------------------------------------- fused attention per (b,h)
__global__ __launch_bounds__(160) void attn_kernel(const __bf16* __restrict__ Qb,
                                                   const __bf16* __restrict__ Kb,
                                                   const __bf16* __restrict__ Vg,
                                                   const float* __restrict__ bias_p,
                                                   __bf16* __restrict__ ctx) {
  __shared__ __bf16 kt_l[SP * DKK];           // K tile   [80][64], 10 KB
  __shared__ __bf16 vt_l[DKK * SPV];          // V^T tile [64][96], 12 KB
  __shared__ __bf16 attn_l[5][16][SPV];       // softmax(QK^T)      15 KB

  const int bh   = blockIdx.x;
  const int b    = bh >> 4;
  const int h    = bh & 15;
  const int lane = threadIdx.x & 31;
  const int wave = threadIdx.x >> 5;
  const int half = lane >> 4;
  const int nc   = lane & 15;
  const int q0   = wave * 16;

  // --- stage K and V^T tiles into LDS (TDM zero-fills the padding) -----------
#if HAVE_TDM
  if (wave == 0) {
    tdm_load_2d(Kb + (size_t)bh * SP * DKK,
                (unsigned)(uintptr_t)(void*)kt_l,
                /*dim0*/ DKK, /*dim1*/ SS, /*til0*/ DKK, /*til1*/ SP,
                /*stride0*/ DKK);
    tdm_load_2d(Vg + (size_t)bh * DKK * SPV,
                (unsigned)(uintptr_t)(void*)vt_l,
                /*dim0*/ SS, /*dim1*/ DKK, /*til0*/ SPV, /*til1*/ DKK,
                /*stride0*/ SPV);
    __builtin_amdgcn_s_wait_tensorcnt(0);
  }
  __syncthreads();
#else
  {
    const bf16x8* ks = (const bf16x8*)(Kb + (size_t)bh * SP * DKK);
    const bf16x8* vs = (const bf16x8*)(Vg + (size_t)bh * DKK * SPV);
    bf16x8* kd = (bf16x8*)kt_l;
    bf16x8* vd = (bf16x8*)vt_l;
    for (int i = threadIdx.x; i < (SP * DKK) / 8; i += 160) kd[i] = ks[i];
    for (int i = threadIdx.x; i < (DKK * SPV) / 8; i += 160) vd[i] = vs[i];
    __syncthreads();
  }
#endif

  // --- scores = (Q K^T) / 8 + bias, then softmax -----------------------------
  int qrow = q0 + nc;
  if (qrow > SS - 1) qrow = SS - 1;            // clamp: padded rows discarded anyway
  const __bf16* Qrow = Qb + ((size_t)bh * SP + qrow) * DKK;
  bf16x16 aq0 = load_a_bf16(Qrow, 0, lane);
  bf16x16 aq1 = load_a_bf16(Qrow, 32, lane);

  f32x8 s[5];
#pragma unroll
  for (int t = 0; t < 5; ++t) {
    const __bf16* Krow = &kt_l[(t * 16 + nc) * DKK + (half << 4)];
    bf16x16 bk0 = *(const bf16x16*)(Krow);
    bf16x16 bk1 = *(const bf16x16*)(Krow + 32);
    s[t] = wmma_bf16(aq1, bk1, wmma_bf16(aq0, bk0, zero8()));
  }

  const float* brow = bias_p + ((size_t)h * SP + q0 + half * 8) * SP + nc;
#pragma unroll
  for (int r = 0; r < 8; ++r) {
    float mx = -1e30f;
#pragma unroll
    for (int t = 0; t < 5; ++t) {
      int k   = t * 16 + nc;
      float v = s[t][r] * 0.125f + brow[r * SP + t * 16];   // 1/sqrt(64), padded bias
      if (k >= SS) v = -1e30f;                              // mask padded keys
      s[t][r] = v;
      mx = fmaxf(mx, v);
    }
#pragma unroll
    for (int off = 8; off; off >>= 1) mx = fmaxf(mx, __shfl_xor(mx, off, 32));
    float sum = 0.0f;
#pragma unroll
    for (int t = 0; t < 5; ++t) {
      float e = __expf(s[t][r] - mx);
      s[t][r] = e;
      sum += e;
    }
#pragma unroll
    for (int off = 8; off; off >>= 1) sum += __shfl_xor(sum, off, 32);
    float inv = 1.0f / sum;
#pragma unroll
    for (int t = 0; t < 5; ++t)
      attn_l[wave][half * 8 + r][t * 16 + nc] = (__bf16)(s[t][r] * inv);
    attn_l[wave][half * 8 + r][80 + nc] = (__bf16)0.0f;     // zero pad cols 80..95
  }

  // per-wave LDS RAW: DS-counter ordering is enough (no cross-wave sharing)
  asm volatile("s_wait_dscnt 0" ::: "memory");

  // --- ctx = attn @ V --------------------------------------------------------
  f32x8 c[4];
#pragma unroll
  for (int dt = 0; dt < 4; ++dt) c[dt] = zero8();

#pragma unroll
  for (int kc = 0; kc < 3; ++kc) {
    int kk = kc * 32;
    bf16x16 aa = load_a_bf16(&attn_l[wave][nc][0], kk, lane);
#pragma unroll
    for (int dt = 0; dt < 4; ++dt) {
      const __bf16* vr = &vt_l[(dt * 16 + nc) * SPV + kk + (half << 4)];
      bf16x16 bv = *(const bf16x16*)vr;
      c[dt] = wmma_bf16(aa, bv, c[dt]);
    }
  }

#pragma unroll
  for (int dt = 0; dt < 4; ++dt) {
#pragma unroll
    for (int r = 0; r < 8; ++r) {
      int q = q0 + half * 8 + r;
      if (q < SS)
        ctx[((size_t)(b * SS + q)) * DD + h * DKK + dt * 16 + nc] = (__bf16)c[dt][r];
    }
  }
}

// ---------------------------------------------------------------- launcher
extern "C" void kernel_launch(void* const* d_in, const int* in_sizes, int n_in,
                              void* d_out, int out_size, void* d_ws, size_t ws_size,
                              hipStream_t stream) {
  const float* q    = (const float*)d_in[0];
  const float* k    = (const float*)d_in[1];
  const float* v    = (const float*)d_in[2];
  // d_in[3] = token_coords (unused by the reference math)
  const float* Wq   = (const float*)d_in[4];
  const float* Wk   = (const float*)d_in[5];
  const float* Wv   = (const float*)d_in[6];
  const float* Wo   = (const float*)d_in[7];
  const float* bias = (const float*)d_in[8];
  float* out = (float*)d_out;

  char* ws = (char*)d_ws;
  __bf16* Wbf = (__bf16*)ws;                                  // 4 * 2 MB
  size_t off = (size_t)4 * DD * DD * sizeof(__bf16);
  __bf16* Qb  = (__bf16*)(ws + off); off += (size_t)BHN * SP  * DKK * 2;  // 80 MB
  __bf16* Kb  = (__bf16*)(ws + off); off += (size_t)BHN * SP  * DKK * 2;  // 80 MB
  __bf16* Vt  = (__bf16*)(ws + off); off += (size_t)BHN * DKK * SPV * 2;  // 96 MB
  __bf16* ctx = (__bf16*)(ws + off); off += (size_t)NTOK * DD * 2;        // 78 MB
  float*  bias_p = (float*)(ws + off);                                    // 0.4 MB

  convert_weights_kernel<<<(4 << 20) / 256, 256, 0, stream>>>(Wq, Wk, Wv, Wo, Wbf);
  pad_bias_kernel<<<(HH * SP * SP + 255) / 256, 256, 0, stream>>>(bias, bias_p);

  dim3 g(DD / 128, NTOK / 128);   // (8, 312)
  proj_gemm<0><<<g, 256, 0, stream>>>(q, nullptr, Wbf,                        Qb, nullptr);
  proj_gemm<0><<<g, 256, 0, stream>>>(k, nullptr, Wbf + (size_t)DD * DD,      Kb, nullptr);
  proj_gemm<1><<<g, 256, 0, stream>>>(v, nullptr, Wbf + (size_t)2 * DD * DD,  Vt, nullptr);

  attn_kernel<<<BHN, 160, 0, stream>>>(Qb, Kb, Vt, bias_p, ctx);

  proj_gemm<2><<<g, 256, 0, stream>>>(nullptr, ctx, Wbf + (size_t)3 * DD * DD,
                                      nullptr, out);
}
